// R2LEncoderBlock_51161650430699
// MI455X (gfx1250) — matrix operand order repaired and checked by reading, not compiled, not run on hardware
//
#include <hip/hip_runtime.h>
#include <hip/hip_bf16.h>
#include <math.h>
#include <stdint.h>

typedef _Float16 half_t;
typedef __attribute__((ext_vector_type(16))) _Float16 v16h;
typedef __attribute__((ext_vector_type(8)))  _Float16 v8h;
typedef __attribute__((ext_vector_type(8)))  float    v8f;

#define N_PTS 32768
#define CH    192
#define NH    6
#define HD    32
#define WK    16

__device__ inline v8f wmma16(v16h a, v16h b, v8f c) {
    return __builtin_amdgcn_wmma_f32_16x16x32_f16(false, a, false, b, (short)0, c,
                                                  false, false);
}

// Async global->LDS copy (ASYNCcnt-tracked). VDST = LDS byte offset VGPR.
__device__ inline void async_load16(const void* gptr, unsigned ldsoff) {
    asm volatile("global_load_async_to_lds_b128 %0, %1, off"
                 :: "v"(ldsoff), "v"(gptr)
                 : "memory");
}
__device__ inline void wait_async0() {
    asm volatile("s_wait_asynccnt 0x0" ::: "memory");
}
__device__ inline unsigned lds_off(const void* p) {
    return (unsigned)(uintptr_t)p;   // LDS aperture: addr[31:0] = LDS offset
}

// ---------------------------------------------------------------------------
// Weight convert + transpose: W (rows x cols, f32) -> WT (cols x rows, f16)
// ---------------------------------------------------------------------------
__global__ void cvt_transpose(const float* __restrict__ W, half_t* __restrict__ WT,
                              int rows, int cols) {
    int i = blockIdx.x * blockDim.x + threadIdx.x;
    int total = rows * cols;
    if (i >= total) return;
    int c = i / rows;
    int r = i - c * rows;
    WT[i] = (half_t)W[r * cols + c];
}

// ---------------------------------------------------------------------------
// xyz column-min reduction (floats >= 0, so uint bit pattern is monotonic)
// ---------------------------------------------------------------------------
__global__ void init_min(unsigned* mb) {
    if (threadIdx.x < 3) mb[threadIdx.x] = 0xFFFFFFFFu;
}

__global__ void min_xyz(const float* __restrict__ xyz, unsigned* __restrict__ mb) {
    __shared__ unsigned sm[3];
    if (threadIdx.x < 3) sm[threadIdx.x] = 0xFFFFFFFFu;
    __syncthreads();
    int stride = gridDim.x * blockDim.x;
    for (int i = blockIdx.x * blockDim.x + threadIdx.x; i < N_PTS * 3; i += stride) {
        atomicMin(&sm[i % 3], __float_as_uint(xyz[i]));
    }
    __syncthreads();
    if (threadIdx.x < 3) atomicMin(&mb[threadIdx.x], sm[threadIdx.x]);
}

__global__ void quant_xyz(const float* __restrict__ xyz,
                          const unsigned* __restrict__ mb, int* __restrict__ xq) {
    int i = blockIdx.x * blockDim.x + threadIdx.x;
    if (i >= N_PTS * 3) return;
    float mn = __uint_as_float(mb[i % 3]);
    xq[i] = (int)floorf(fmodf(xyz[i] - mn, 8.0f));  // WIN=8, QUANT=1, shift=0
}

// ---------------------------------------------------------------------------
// LayerNorm (C=192) -> f16 output. One wave per row, 6 elems per lane.
// ---------------------------------------------------------------------------
__global__ __launch_bounds__(256) void ln_to_half(const float* __restrict__ x,
                                                  const float* __restrict__ g,
                                                  const float* __restrict__ b,
                                                  half_t* __restrict__ out, int n) {
    int wave = threadIdx.x >> 5;
    int lane = threadIdx.x & 31;
    int row  = blockIdx.x * 8 + wave;
    if (row >= n) return;
    const float* xr = x + (size_t)row * CH;
    float v[6];
    float s = 0.f;
#pragma unroll
    for (int k = 0; k < 6; ++k) { v[k] = xr[lane + 32 * k]; s += v[k]; }
#pragma unroll
    for (int off = 16; off >= 1; off >>= 1) s += __shfl_xor(s, off, 32);
    float mean = s * (1.0f / CH);
    float vs = 0.f;
#pragma unroll
    for (int k = 0; k < 6; ++k) { float d = v[k] - mean; vs += d * d; }
#pragma unroll
    for (int off = 16; off >= 1; off >>= 1) vs += __shfl_xor(vs, off, 32);
    float rstd = rsqrtf(vs * (1.0f / CH) + 1e-5f);
    half_t* orow = out + (size_t)row * CH;
#pragma unroll
    for (int k = 0; k < 6; ++k) {
        int c = lane + 32 * k;
        orow[c] = (half_t)((v[k] - mean) * rstd * g[c] + b[c]);
    }
}

// ---------------------------------------------------------------------------
// WMMA GEMM: out[M x Ncols] = A_h[M x KC] @ BT_h[Ncols x KC]^T + bias [+ epi]
// grid = (M/32, Ncols/192), block = 128 (4 waves; 2 row-tiles x 3 col-tiles).
// A K-slabs double-buffered in LDS via global_load_async_to_lds_b128.
// EPI: 0 = bias -> f32 ; 1 = bias + residual -> f32 ; 2 = bias + GELU -> f16
// ---------------------------------------------------------------------------
template <int EPI, int KC>
__global__ __launch_bounds__(128) void gemm_wmma(const half_t* __restrict__ A,
                                                 const half_t* __restrict__ BT,
                                                 const float* __restrict__ bias,
                                                 const float* __restrict__ resid,
                                                 float* __restrict__ outF,
                                                 half_t* __restrict__ outH,
                                                 int Ncols) {
    __shared__ alignas(32) half_t At[2][32][32];   // double-buffered 32x32 K-slab
    int tid  = threadIdx.x;
    int wave = tid >> 5, lane = tid & 31;
    int hf   = lane >> 4, l16 = lane & 15;
    int row0    = blockIdx.x * 32;
    int colbase = blockIdx.y * 192 + wave * 48;

    // cooperative async A-slab copy: one b128 (8 halves) per thread
    int ldrow = tid >> 2;
    int ldcol = (tid & 3) * 8;
    const half_t* agp = &A[(size_t)(row0 + ldrow) * KC + ldcol];
    unsigned lbase = lds_off(&At[0][ldrow][ldcol]);

    v8f acc[2][3] = {};

    constexpr int NK = KC / 32;
    async_load16(agp, lbase);                       // kt=0 -> buffer 0
    wait_async0();
    __syncthreads();

#pragma unroll
    for (int kt = 0; kt < NK; ++kt) {
        int buf = kt & 1;
        if (kt + 1 < NK)                            // prefetch next slab into buf^1
            async_load16(agp + (kt + 1) * 32, lbase + ((kt + 1) & 1) * 2048);

        // A fragments (contiguous K: lane-half K = 0-15 / 16-31, matches B)
        v16h af[2];
#pragma unroll
        for (int rt = 0; rt < 2; ++rt) {
            v8h lo = *(const v8h*)&At[buf][rt * 16 + l16][hf * 16];
            v8h hi = *(const v8h*)&At[buf][rt * 16 + l16][hf * 16 + 8];
#pragma unroll
            for (int i = 0; i < 8; ++i) { af[rt][i] = lo[i]; af[rt][i + 8] = hi[i]; }
        }

        if (kt + 1 < NK)   // keep a global_prefetch_b8 on the B stream too
            __builtin_prefetch(&BT[(size_t)(colbase + l16) * KC + (kt + 1) * 32], 0, 1);

#pragma unroll
        for (int t = 0; t < 3; ++t) {
            // B fragment: lane holds col=l16, 16 contiguous K at hf*16
            const half_t* bp =
                &BT[(size_t)(colbase + t * 16 + l16) * KC + kt * 32 + hf * 16];
            v16h bf = *(const v16h*)bp;
            acc[0][t] = wmma16(af[0], bf, acc[0][t]);
            acc[1][t] = wmma16(af[1], bf, acc[1][t]);
        }
        wait_async0();
        __syncthreads();
    }

#pragma unroll
    for (int rt = 0; rt < 2; ++rt) {
#pragma unroll
        for (int t = 0; t < 3; ++t) {
            int col = colbase + t * 16 + l16;
            float bv = bias[col];
#pragma unroll
            for (int r = 0; r < 8; ++r) {
                int m = rt * 16 + hf * 8 + r;   // C-layout: lane half -> M 0-7 / 8-15
                size_t o = (size_t)(row0 + m) * Ncols + col;
                float v = acc[rt][t][r] + bv;
                if (EPI == 1) v += resid[o];
                if (EPI == 2) {
                    v = 0.5f * v * (1.0f + erff(v * 0.70710678118654752f));
                    outH[o] = (half_t)v;
                } else {
                    outF[o] = v;
                }
            }
        }
    }
}

// ---------------------------------------------------------------------------
// Windowed attention: one block = one 16-point window, one wave per head.
//   S = (Q*s) K^T                       (1 WMMA)
//   D_c[i][rel] = (Q*s) . T_c[rel]      (3 WMMAs, rel-pos bias as GEMM)
//   bias[i][j] = sum_c D_c[i][rel_ijc]  (3 LDS gathers / element)
//   softmax; O = P V                    (2 WMMAs, K padded 16->32)
// Per-head LDS scratch sWS: holds D (3*16*16), then aliased by P (16*17).
// ---------------------------------------------------------------------------
__global__ __launch_bounds__(192) void attn_wmma(const float* __restrict__ qkv,
                                                 const int* __restrict__ xq,
                                                 const float* __restrict__ table,
                                                 half_t* __restrict__ outH) {
    __shared__ alignas(16) float sQKV[WK * 576];   // 16 pts x (3*H*D)   36.0 KB
    __shared__ float sWS[NH][768];                 // per-head D/P scratch 18.0 KB
    __shared__ int sXQ[WK][3];

    int w = blockIdx.x;
    int tid = threadIdx.x;

    // async stage of the 36864B QKV window: 12 x b128 per thread
    {
        const float* src = qkv + (size_t)w * WK * 576;
#pragma unroll
        for (int k = 0; k < 12; ++k) {
            int i = (tid + k * 192) * 4;                 // float index
            async_load16(src + i, lds_off(&sQKV[i]));
        }
    }
    if (tid < 48) ((int*)sXQ)[tid] = xq[w * 48 + tid];
    wait_async0();
    __syncthreads();

    int h = tid >> 5, lane = tid & 31;
    int hf = lane >> 4, l16 = lane & 15;
    const float qscale = 0.17677669529663687f;  // D^-0.5, D=32

    // Q fragment (A layout, scaled), K fragment (B layout, contiguous K)
    v16h qf, kf;
    int qrow = l16 * 576 + h * 32;               // q of point l16, head h
    int krow = l16 * 576 + 192 + h * 32;         // k of point l16, head h
#pragma unroll
    for (int i = 0; i < 8; ++i) {
        qf[i]     = (half_t)(sQKV[qrow + hf * 16 + i] * qscale);
        qf[i + 8] = (half_t)(sQKV[qrow + hf * 16 + 8 + i] * qscale);
    }
#pragma unroll
    for (int i = 0; i < 16; ++i) kf[i] = (half_t)sQKV[krow + hf * 16 + i];

    v8f s = {};
    s = wmma16(qf, kf, s);   // S[i][j], lane holds j=l16, rows i = hf*8 + r

    // rel-pos bias as GEMM: D_c[i][rel] = (Q*s) . T_c[rel], rel = 0..14 (+pad)
    int relrow = (l16 < 15) ? l16 : 0;           // clamp pad lane (no OOB load)
#pragma unroll
    for (int c = 0; c < 3; ++c) {
        const float* tb = table + ((size_t)(relrow * NH + h) * HD) * 3 + c;
        v16h tf;
#pragma unroll
        for (int i = 0; i < 16; ++i) {
            float tv = tb[(hf * 16 + i) * 3];    // T[rel][h][d][c], d = hf*16+i
            tf[i] = (l16 < 15) ? (half_t)tv : (half_t)0.f;
        }
        v8f dacc = {};
        dacc = wmma16(qf, tf, dacc);             // lane: col rel=l16, rows hf*8+r
#pragma unroll
        for (int r = 0; r < 8; ++r)
            sWS[h][(c * 16 + hf * 8 + r) * 16 + l16] = dacc[r];
    }

    // gather bias: bias[i][j] = sum_c D_c[i][xq_i[c]-xq_j[c]+7]
    int j = l16;
#pragma unroll
    for (int r = 0; r < 8; ++r) {
        int i = hf * 8 + r;
        float b = 0.f;
#pragma unroll
        for (int c = 0; c < 3; ++c) {
            int rel = sXQ[i][c] - sXQ[j][c] + 7;  // L-1 = 7, in [0,14]
            b += sWS[h][(c * 16 + i) * 16 + rel];
        }
        s[r] += b;
    }

    // softmax over keys j (16 lanes within each half-wave group)
#pragma unroll
    for (int r = 0; r < 8; ++r) {
        float m = s[r];
#pragma unroll
        for (int off = 8; off >= 1; off >>= 1) m = fmaxf(m, __shfl_xor(m, off, 32));
        float e = expf(s[r] - m);
        float sum = e;
#pragma unroll
        for (int off = 8; off >= 1; off >>= 1) sum += __shfl_xor(sum, off, 32);
        sWS[h][(hf * 8 + r) * 17 + l16] = e / sum;   // P aliases dead D scratch
    }
    __syncthreads();

    // O = P(16x16) @ V(16x32): pad K 16->32 with zeros, two 16-col WMMAs
    v16h pf;
#pragma unroll
    for (int i = 0; i < 8; ++i) {
        pf[i]     = (half_t)sWS[h][l16 * 17 + hf * 8 + i];  // K = hf*8+i (< 16)
        pf[i + 8] = (half_t)0.f;                            // K = 16.. : zero pad
    }
#pragma unroll
    for (int t = 0; t < 2; ++t) {
        int d = t * 16 + l16;
        v16h vf;
#pragma unroll
        for (int i = 0; i < 16; ++i) {
            // B layout: lane col = d, K = hf*8 + i for i < 8 (keys >= 16 are zero pad)
            vf[i] = (i < 8) ? (half_t)sQKV[(hf * 8 + i) * 576 + 384 + h * 32 + d]
                            : (half_t)0.f;
        }
        v8f o = {};
        o = wmma16(pf, vf, o);
#pragma unroll
        for (int r = 0; r < 8; ++r) {
            int i = hf * 8 + r;
            outH[(size_t)(w * WK + i) * CH + h * 32 + d] = (half_t)o[r];
        }
    }
}

// ---------------------------------------------------------------------------
extern "C" void kernel_launch(void* const* d_in, const int* in_sizes, int n_in,
                              void* d_out, int out_size, void* d_ws, size_t ws_size,
                              hipStream_t stream) {
    const float* feats  = (const float*)d_in[0];
    const float* xyz    = (const float*)d_in[1];
    const float* n1g    = (const float*)d_in[5];
    const float* n1b    = (const float*)d_in[6];
    const float* qkv_w  = (const float*)d_in[7];
    const float* qkv_b  = (const float*)d_in[8];
    const float* table  = (const float*)d_in[9];
    const float* proj_w = (const float*)d_in[10];
    const float* proj_b = (const float*)d_in[11];
    const float* n2g    = (const float*)d_in[12];
    const float* n2b    = (const float*)d_in[13];
    const float* fc1_w  = (const float*)d_in[14];
    const float* fc1_b  = (const float*)d_in[15];
    const float* fc2_w  = (const float*)d_in[16];
    const float* fc2_b  = (const float*)d_in[17];

    char* ws = (char*)d_ws;
    size_t o = 0;
    auto take = [&](size_t bytes) {
        char* p = ws + o;
        o = (o + bytes + 255) & ~(size_t)255;
        return p;
    };
    half_t*   qkvT = (half_t*)take((size_t)576 * 192 * 2);
    half_t*   projT= (half_t*)take((size_t)192 * 192 * 2);
    half_t*   fc1T = (half_t*)take((size_t)768 * 192 * 2);
    half_t*   fc2T = (half_t*)take((size_t)192 * 768 * 2);
    half_t*   xln  = (half_t*)take((size_t)N_PTS * 192 * 2);
    float*    qkvF = (float*) take((size_t)N_PTS * 576 * 4);
    unsigned* minb = (unsigned*)take(256);
    int*      xq   = (int*)   take((size_t)N_PTS * 3 * 4);
    half_t*   aout = (half_t*)take((size_t)N_PTS * 192 * 2);
    float*    x2   = (float*) take((size_t)N_PTS * 192 * 4);
    half_t*   x2ln = (half_t*)take((size_t)N_PTS * 192 * 2);
    half_t*   h1   = (half_t*)take((size_t)N_PTS * 768 * 2);
    float*    outF = (float*)d_out;

    // weight conversion (f32 row-major -> f16 transposed)
    cvt_transpose<<<(192 * 576 + 255) / 256, 256, 0, stream>>>(qkv_w, qkvT, 192, 576);
    cvt_transpose<<<(192 * 192 + 255) / 256, 256, 0, stream>>>(proj_w, projT, 192, 192);
    cvt_transpose<<<(192 * 768 + 255) / 256, 256, 0, stream>>>(fc1_w, fc1T, 192, 768);
    cvt_transpose<<<(768 * 192 + 255) / 256, 256, 0, stream>>>(fc2_w, fc2T, 768, 192);

    // quantized coordinates
    init_min<<<1, 32, 0, stream>>>(minb);
    min_xyz<<<128, 256, 0, stream>>>(xyz, minb);
    quant_xyz<<<(N_PTS * 3 + 255) / 256, 256, 0, stream>>>(xyz, minb, xq);

    // LN1 -> qkv GEMM
    ln_to_half<<<N_PTS / 8, 256, 0, stream>>>(feats, n1g, n1b, xln, N_PTS);
    gemm_wmma<0, 192><<<dim3(N_PTS / 32, 3), 128, 0, stream>>>(
        xln, qkvT, qkv_b, nullptr, qkvF, nullptr, 576);

    // windowed attention
    attn_wmma<<<N_PTS / WK, 192, 0, stream>>>(qkvF, xq, table, aout);

    // proj GEMM + residual(feats) -> x2
    gemm_wmma<1, 192><<<dim3(N_PTS / 32, 1), 128, 0, stream>>>(
        aout, projT, proj_b, feats, x2, nullptr, 192);

    // LN2 -> fc1 GEMM (+GELU, f16) -> fc2 GEMM (+residual x2) -> out
    ln_to_half<<<N_PTS / 8, 256, 0, stream>>>(x2, n2g, n2b, x2ln, N_PTS);
    gemm_wmma<2, 192><<<dim3(N_PTS / 32, 4), 128, 0, stream>>>(
        x2ln, fc1T, fc1_b, nullptr, nullptr, h1, 768);
    gemm_wmma<1, 768><<<dim3(N_PTS / 32, 1), 128, 0, stream>>>(
        h1, fc2T, fc2_b, x2, outF, nullptr, 192);
}
